// Attention_171798692167
// MI455X (gfx1250) — matrix operand order repaired and checked by reading, not compiled, not run on hardware
//
#include <hip/hip_runtime.h>

// ---------------------------------------------------------------------------
// Types for CDNA5 WMMA (wave32, 16x16x32 bf16 -> f32)
// ---------------------------------------------------------------------------
typedef __bf16 bf16_t;
typedef bf16_t bf16x4  __attribute__((ext_vector_type(4)));
typedef bf16_t bf16x8  __attribute__((ext_vector_type(8)));
typedef bf16_t bf16x16 __attribute__((ext_vector_type(16)));
typedef float  floatx8 __attribute__((ext_vector_type(8)));

#define WMMA_BF16 __builtin_amdgcn_wmma_f32_16x16x32_bf16

// Hardware tanh (V_TANH_F32 is a CDNA5 TRANS32 op). Guarded so compile never
// breaks if the builtin spelling differs in this toolchain.
#if defined(__has_builtin)
#  if __has_builtin(__builtin_amdgcn_tanhf)
#    define FAST_TANH(x) __builtin_amdgcn_tanhf(x)
#  elif __has_builtin(__builtin_amdgcn_tanh_f32)
#    define FAST_TANH(x) __builtin_amdgcn_tanh_f32(x)
#  else
#    define FAST_TANH(x) tanhf(x)
#  endif
#else
#  define FAST_TANH(x) tanhf(x)
#endif

// Problem constants
#define B_  128
#define N_  2048
#define D_  512     // DM == DQ == DA == 512
#define TM_ 64      // rows of N per workgroup in the logits kernel

// ---------------------------------------------------------------------------
// Kernel 0a: pack Wm (f32, [D][D] row-major, k-major) into bf16 in the exact
// B-fragment order.  B-matrix (32x16, 16-bit) layout per ISA 7.12.2:
//   lane l holds column N = l&15; lanes 0-15 hold K=k0..k0+7, k0+16..k0+23 (elements 0..15),
//   lanes 16-31 hold K=k0+8..k0+15, k0+24..k0+31.
// Pack address: ((((t*16 + s)*32) + l)*16 + f) with f = element 0..15, so a
// wave's B fragment is a single contiguous, coalesced 32B load per lane.
// ---------------------------------------------------------------------------
__global__ __launch_bounds__(256) void bpack_kernel(const float* __restrict__ Wm,
                                                    bf16_t* __restrict__ Bp) {
    int o = blockIdx.x * 256 + threadIdx.x;        // 0 .. 512*512-1
    int f = o & 15;                                // element within fragment
    int l = (o >> 4) & 31;                         // lane
    int s = (o >> 9) & 15;                         // k-step (K block of 32)
    int t = o >> 13;                               // a-tile (0..31)
    int a = t * 16 + (l & 15);
    int k = s * 32 + ((l < 16) ? 0 : 8) + (f < 8 ? f : f + 8);
    Bp[o] = (bf16_t)Wm[k * D_ + a];
}

// ---------------------------------------------------------------------------
// Kernel 0b: tq = query @ Wq   (B x D)
// ---------------------------------------------------------------------------
__global__ __launch_bounds__(256) void tq_kernel(const float* __restrict__ q,
                                                 const float* __restrict__ Wq,
                                                 float* __restrict__ tq) {
    int idx = blockIdx.x * 256 + threadIdx.x;      // 0 .. B*D-1
    int b = idx >> 9;
    int a = idx & (D_ - 1);
    const float* qb = q + (size_t)b * D_;
    float s = 0.f;
    for (int d = 0; d < D_; ++d) s = fmaf(qb[d], Wq[d * D_ + a], s);
    tq[idx] = s;
}

// ---------------------------------------------------------------------------
// Kernel 1: logits_raw[b, n] = tanh(H[b,n,:] @ Wm + tq[b,:]) . v
// One 256-thread block (8 waves) per (b, 64-row tile of N).
// Wave w: m-subtiles 0..3 (rows), a-tiles {w, w+8, w+16, w+24} in 2 passes.
// ---------------------------------------------------------------------------
__global__ __launch_bounds__(256) void logits_kernel(const float* __restrict__ H,
                                                     const bf16_t* __restrict__ Bp,
                                                     const float* __restrict__ tq,
                                                     const float* __restrict__ vv,
                                                     float* __restrict__ logits_raw) {
    __shared__ bf16_t Hs[TM_][D_ + 8];   // +8 bf16 pad -> 260-DWORD row stride (no bank conflicts)
    __shared__ float  wlog[8][TM_];      // per-wave partial logits (deterministic reduce)

    const int tid  = threadIdx.x;
    const int b    = blockIdx.y;
    const int nt   = blockIdx.x;
    const int wave = tid >> 5;
    const int lane = tid & 31;

    // zero per-wave partials
    for (int i = tid; i < 8 * TM_; i += 256) (&wlog[0][0])[i] = 0.f;

    // Stage H tile (64 x 512 f32, contiguous) -> bf16 LDS, coalesced b128 loads
    const float* hsrc = H + ((size_t)b * N_ + (size_t)nt * TM_) * D_;
    const float4* hsrc4 = (const float4*)hsrc;
    for (int i = 0; i < 32; ++i) {
        int idx4 = tid + i * 256;          // 0..8191 float4 units
        int row  = idx4 >> 7;              // / (512/4)
        int c4   = idx4 & 127;
        float4 fv = hsrc4[idx4];
        bf16x4 hv = { (bf16_t)fv.x, (bf16_t)fv.y, (bf16_t)fv.z, (bf16_t)fv.w };
        *(bf16x4*)&Hs[row][c4 * 4] = hv;
    }
    __syncthreads();

    const int koff  = (lane < 16) ? 0 : 8;   // A-fragment K offset (16-bit A layout)
    const int rowin = lane & 15;

    for (int jj = 0; jj < 2; ++jj) {         // a-tile pass (2 a-tiles per pass)
        floatx8 acc[2][4];
        #pragma unroll
        for (int jd = 0; jd < 2; ++jd)
            #pragma unroll
            for (int m = 0; m < 4; ++m)
                acc[jd][m] = (floatx8)0.f;

        for (int s = 0; s < 16; ++s) {       // K loop, 32 per step
            bf16x16 afrag[4];
            #pragma unroll
            for (int m = 0; m < 4; ++m) {
                const bf16_t* p = &Hs[m * 16 + rowin][s * 32 + koff];
                bf16x8 lo = *(const bf16x8*)p;          // K = k0+koff .. +7
                bf16x8 hi = *(const bf16x8*)(p + 16);   // K = k0+koff+16 .. +23
                afrag[m] = __builtin_shufflevector(lo, hi,
                    0, 1, 2, 3, 4, 5, 6, 7, 8, 9, 10, 11, 12, 13, 14, 15);
            }
            #pragma unroll
            for (int jd = 0; jd < 2; ++jd) {
                int t = wave + (jj * 2 + jd) * 8;       // a-tile 0..31
                const bf16x16 bfrag =
                    *(const bf16x16*)(Bp + ((size_t)((t * 16 + s) * 32 + lane) << 4));
                #pragma unroll
                for (int m = 0; m < 4; ++m)
                    acc[jd][m] = WMMA_BF16(false, afrag[m], false, bfrag,
                                           (short)0, acc[jd][m], false, false);
            }
        }

        // Epilogue: e = tanh(acc + tq[a]) * v[a], reduce over the 16 a-columns
        float lacc[4][8];
        #pragma unroll
        for (int m = 0; m < 4; ++m)
            #pragma unroll
            for (int r = 0; r < 8; ++r) lacc[m][r] = 0.f;

        #pragma unroll
        for (int jd = 0; jd < 2; ++jd) {
            int t = wave + (jj * 2 + jd) * 8;
            int a = t * 16 + rowin;
            float tqa = tq[b * D_ + a];
            float va  = vv[a];
            #pragma unroll
            for (int m = 0; m < 4; ++m)
                #pragma unroll
                for (int r = 0; r < 8; ++r)
                    lacc[m][r] += FAST_TANH(acc[jd][m][r] + tqa) * va;
        }

        // Butterfly-sum within each 16-lane half (lanes share rows, differ in a)
        #pragma unroll
        for (int m = 0; m < 4; ++m)
            #pragma unroll
            for (int r = 0; r < 8; ++r) {
                float x = lacc[m][r];
                x += __shfl_xor(x, 1);
                x += __shfl_xor(x, 2);
                x += __shfl_xor(x, 4);
                x += __shfl_xor(x, 8);
                int id = m * 8 + r;
                if ((lane & 15) == (id & 15))
                    wlog[wave][m * 16 + r + ((lane < 16) ? 0 : 8)] += x;
            }
    }

    __syncthreads();
    if (tid < TM_) {
        float s = 0.f;
        #pragma unroll
        for (int w = 0; w < 8; ++w) s += wlog[w][tid];
        logits_raw[(size_t)b * N_ + (size_t)nt * TM_ + tid] = s;
    }
}

// ---------------------------------------------------------------------------
// Kernel 2: masked softmax per row; writes weights and (suppressed) logits
// ---------------------------------------------------------------------------
__global__ __launch_bounds__(256) void softmax_kernel(const float* __restrict__ lraw,
                                                      const float* __restrict__ mask,
                                                      float* __restrict__ w_out,
                                                      float* __restrict__ l_out) {
    __shared__ float red[256];
    const int tid = threadIdx.x;
    const int b   = blockIdx.x;
    const float* lr = lraw + (size_t)b * N_;
    const float* mk = mask + (size_t)b * N_;

    // any valid cell? (mask is 0/1 float)
    float mv = 0.f;
    for (int n = tid; n < N_; n += 256) mv = fmaxf(mv, mk[n]);
    red[tid] = mv; __syncthreads();
    for (int st = 128; st > 0; st >>= 1) {
        if (tid < st) red[tid] = fmaxf(red[tid], red[tid + st]);
        __syncthreads();
    }
    const bool any = red[0] > 0.f;
    __syncthreads();

    float lv[8];
    float lmax = -3.4e38f;
    #pragma unroll
    for (int i = 0; i < 8; ++i) {
        int n = tid + i * 256;
        float x = lr[n];
        if (any && mk[n] == 0.f) x += -1e9f;   // suppress
        lv[i] = x;
        lmax = fmaxf(lmax, x);
        l_out[(size_t)b * N_ + n] = x;
    }
    red[tid] = lmax; __syncthreads();
    for (int st = 128; st > 0; st >>= 1) {
        if (tid < st) red[tid] = fmaxf(red[tid], red[tid + st]);
        __syncthreads();
    }
    const float gmax = red[0];
    __syncthreads();

    float ev[8];
    float ls = 0.f;
    #pragma unroll
    for (int i = 0; i < 8; ++i) { ev[i] = expf(lv[i] - gmax); ls += ev[i]; }
    red[tid] = ls; __syncthreads();
    for (int st = 128; st > 0; st >>= 1) {
        if (tid < st) red[tid] = red[tid] + red[tid + st];
        __syncthreads();
    }
    const float inv = any ? (1.f / red[0]) : 0.f;
    #pragma unroll
    for (int i = 0; i < 8; ++i)
        w_out[(size_t)b * N_ + tid + i * 256] = ev[i] * inv;
}

// ---------------------------------------------------------------------------
// Kernel 3: context partials  part[b][s][d] = sum_{n in seg s} w[b,n]*H[b,n,d]
// ---------------------------------------------------------------------------
__global__ __launch_bounds__(256) void ctx_partial_kernel(const float* __restrict__ H,
                                                          const float* __restrict__ w,
                                                          float* __restrict__ part) {
    const int s = blockIdx.x;   // 0..7
    const int b = blockIdx.y;
    const int tid = threadIdx.x;
    const float* wrow = w + (size_t)b * N_ + s * 256;
    const float2* Hp  = (const float2*)(H + ((size_t)b * N_ + (size_t)s * 256) * D_);
    float ax = 0.f, ay = 0.f;
    for (int n = 0; n < 256; ++n) {
        float wn = wrow[n];
        float2 hv = Hp[(size_t)n * 256 + tid];
        ax = fmaf(wn, hv.x, ax);
        ay = fmaf(wn, hv.y, ay);
    }
    float2 r; r.x = ax; r.y = ay;
    ((float2*)(part + ((size_t)(b * 8 + s)) * D_))[tid] = r;
}

__global__ __launch_bounds__(256) void ctx_reduce_kernel(const float* __restrict__ part,
                                                         float* __restrict__ ctx) {
    int o = blockIdx.x * 256 + threadIdx.x;   // 0 .. B*D-1
    int b = o >> 9;
    int d = o & (D_ - 1);
    float s = 0.f;
    #pragma unroll
    for (int j = 0; j < 8; ++j) s += part[(size_t)((b * 8 + j) << 9) + d];
    ctx[o] = s;
}

// ---------------------------------------------------------------------------
// Launch
// ---------------------------------------------------------------------------
extern "C" void kernel_launch(void* const* d_in, const int* in_sizes, int n_in,
                              void* d_out, int out_size, void* d_ws, size_t ws_size,
                              hipStream_t stream) {
    const float* H     = (const float*)d_in[0];   // (B, N, DM)
    const float* mask  = (const float*)d_in[1];   // (B, N)
    const float* query = (const float*)d_in[2];   // (B, DQ)
    const float* Wm    = (const float*)d_in[3];   // (DM, DA)
    const float* Wq    = (const float*)d_in[4];   // (DQ, DA)
    const float* vvec  = (const float*)d_in[5];   // (DA, 1)

    float* out     = (float*)d_out;
    float* w_out   = out;                                  // B*N
    float* ctx_out = out + (size_t)B_ * N_;                // B*DM
    float* l_out   = ctx_out + (size_t)B_ * D_;            // B*N

    char* ws = (char*)d_ws;
    bf16_t* Bp   = (bf16_t*)ws;                            // 512 KB
    float*  tq   = (float*)(ws + (512u << 10));            // 256 KB
    float*  lraw = (float*)(ws + (768u << 10));            // 1 MB
    float*  part = (float*)(ws + (1792u << 10));           // 2 MB

    bpack_kernel<<<(D_ * D_) / 256, 256, 0, stream>>>(Wm, Bp);
    tq_kernel<<<(B_ * D_) / 256, 256, 0, stream>>>(query, Wq, tq);
    logits_kernel<<<dim3(N_ / TM_, B_), 256, 0, stream>>>(H, Bp, tq, vvec, lraw);
    softmax_kernel<<<B_, 256, 0, stream>>>(lraw, mask, w_out, l_out);
    ctx_partial_kernel<<<dim3(8, B_), 256, 0, stream>>>(H, w_out, part);
    ctx_reduce_kernel<<<(B_ * D_) / 256, 256, 0, stream>>>(part, ctx_out);
}
